// SemanticHead_90778428768742
// MI455X (gfx1250) — matrix operand order repaired and checked by reading, not compiled, not run on hardware
//
#include <hip/hip_runtime.h>
#include <hip/hip_bf16.h>
#include <stdint.h>

// ---- CDNA5 WMMA types (per CDNA5_HIP.md, probe-confirmed) ----
typedef __attribute__((ext_vector_type(16))) __bf16 v16bf;
typedef __attribute__((ext_vector_type(8)))  __bf16 bf16x8;
typedef __attribute__((ext_vector_type(4)))  __bf16 bf16x4;
typedef __attribute__((ext_vector_type(8)))  float  v8f;

union ABf { v16bf v; bf16x8 h[2]; };

// ---- Problem constants (fixed by reference) ----
#define NPIX      262144      // B*H*W = 16*128*128
#define C_DIM     384
#define D_DIM     64
#define K_CL      27
#define ROWS_PB   64          // pixels per block
#define XN_STRIDE 392         // bf16 elems; 784B = 196 DW -> 4-bank row skew
#define S_STRIDE  68          // f32 elems;  272B row skew

// ---- workspace layout (bytes) ----
#define WS_W1 0               // 384*384 bf16 = 294912 B
#define WS_WL 294912          // 64*384  bf16 = 49152 B
#define WS_W2 344064          // 64*384  bf16 = 49152 B
#define WS_NC 393216          // 27*64   f32  = 6912 B

// ---- dynamic LDS layout (bytes) ----
#define L_XN 0                // 64*392*2 = 50176
#define L_Y1 50176            // 64*392*2 = 50176
#define L_S  100352           // 64*68*4  = 17408
#define L_WB 117760           // 2*16*392*2 = 25088 (weight staging, both N-halves)
#define L_PS 142848           // 64*4*4   = 1024
#define L_RN 143872           // 64*4     = 256
#define L_TOTAL 144128

// ---- CDNA5 async global->LDS staging (ASYNCcnt path), guarded ----
typedef int v4i_vs __attribute__((vector_size(16)));

#if defined(__gfx1250__) && __has_builtin(__builtin_amdgcn_global_load_async_to_lds_b128)
#define ASYNC_STAGE 1
static __device__ __forceinline__ void stage_cp16(const __bf16* g, __bf16* l) {
    // Param types per hipcc diagnostic: AS1 pointer-to-int4 source, AS3 dest.
    __builtin_amdgcn_global_load_async_to_lds_b128(
        (__attribute__((address_space(1))) v4i_vs*)(uintptr_t)g,
        (__attribute__((address_space(3))) v4i_vs*)l,
        0, 0);
}
static __device__ __forceinline__ void stage_wait() {
#if __has_builtin(__builtin_amdgcn_s_wait_asynccnt)
    __builtin_amdgcn_s_wait_asynccnt(0);
#else
    asm volatile("s_wait_asynccnt 0x0" ::: "memory");
#endif
}
#else
#define ASYNC_STAGE 0
static __device__ __forceinline__ void stage_cp16(const __bf16* g, __bf16* l) {
    *(bf16x8*)l = *(const bf16x8*)g;
}
static __device__ __forceinline__ void stage_wait() {}
#endif

// ============================================================
// Kernel 0: weights f32 -> bf16, centers -> normalized f32
// ============================================================
__global__ void prep_kernel(const float* __restrict__ w1,
                            const float* __restrict__ wl,
                            const float* __restrict__ w2,
                            const float* __restrict__ cc,
                            __bf16* __restrict__ w1b,
                            __bf16* __restrict__ wlb,
                            __bf16* __restrict__ w2b,
                            float*  __restrict__ nc) {
    const int b = blockIdx.x;
    if (b < 768) {
        const int i = b * 256 + threadIdx.x;   // exactly covers 196608 elems
        if (i < 147456) {
            w1b[i] = (__bf16)w1[i];
        } else if (i < 172032) {
            wlb[i - 147456] = (__bf16)wl[i - 147456];
        } else {
            w2b[i - 172032] = (__bf16)w2[i - 172032];
        }
    } else {
        const int k = threadIdx.x;
        if (k < K_CL) {
            float s = 0.f;
            for (int d = 0; d < D_DIM; d++) { float v = cc[k * D_DIM + d]; s += v * v; }
            const float inv = 1.0f / fmaxf(sqrtf(s), 1e-10f);
            for (int d = 0; d < D_DIM; d++) nc[k * D_DIM + d] = cc[k * D_DIM + d] * inv;
        }
    }
}

// ============================================================
// Main fused kernel: 256 threads (8 wave32), 64 pixels/block
// ============================================================
__global__ void __launch_bounds__(256)
semantic_head_kernel(const float* __restrict__ features,
                     const float* __restrict__ b_lin,
                     const float* __restrict__ b_nl1,
                     const float* __restrict__ b_nl2,
                     const __bf16* __restrict__ w1b,
                     const __bf16* __restrict__ wlb,
                     const __bf16* __restrict__ w2b,
                     const float* __restrict__ ncf,
                     const int* __restrict__ pseudo,
                     int* __restrict__ out) {
    extern __shared__ char smem[];
    __bf16* Xn   = (__bf16*)(smem + L_XN);
    __bf16* Y1   = (__bf16*)(smem + L_Y1);
    float*  S    = (float*)(smem + L_S);
    __bf16* wbuf = (__bf16*)(smem + L_WB);
    float*  psum = (float*)(smem + L_PS);
    float*  rn   = (float*)(smem + L_RN);

    const int tid  = threadIdx.x;
    const int lane = tid & 31;
    const int wv   = tid >> 5;
    const int lh   = lane >> 4;     // lane half: K/M sub-block selector
    const int l15  = lane & 15;
    const int mrow = (wv >> 1) * 16; // 4 wave-pairs x 16 rows = 64 rows
    const int nh   = wv & 1;         // N-half within the wave pair

    // -------- Phase A: row L2-normalize -> bf16 tile in LDS --------
    {
        const int r = tid >> 2;          // 0..63
        const int q = tid & 3;           // column quarter (96 f32 each)
        const float* frow = features + ((size_t)blockIdx.x * ROWS_PB + r) * C_DIM + q * 96;
        float s = 0.f;
        #pragma unroll
        for (int i = 0; i < 24; i++) {
            float4 v = ((const float4*)frow)[i];
            s += v.x * v.x + v.y * v.y + v.z * v.z + v.w * v.w;
        }
        psum[r * 4 + q] = s;
        __syncthreads();
        if (tid < ROWS_PB) {
            float t = psum[tid * 4] + psum[tid * 4 + 1] + psum[tid * 4 + 2] + psum[tid * 4 + 3];
            rn[tid] = 1.0f / fmaxf(sqrtf(t), 1e-10f);
        }
        __syncthreads();
        const float scale = rn[r];
        __bf16* xrow = Xn + r * XN_STRIDE + q * 96;
        #pragma unroll
        for (int i = 0; i < 24; i++) {
            float4 v = ((const float4*)frow)[i];
            bf16x4 o;
            o.x = (__bf16)(v.x * scale);
            o.y = (__bf16)(v.y * scale);
            o.z = (__bf16)(v.z * scale);
            o.w = (__bf16)(v.w * scale);
            *(bf16x4*)(xrow + i * 4) = o;
        }
        __syncthreads();
    }

    // -------- Phase B: Y1 = relu(Xn * W1^T + b1), bf16 in LDS --------
    for (int nt = 0; nt < 12; nt++) {
        // stage W1 rows for both N-halves: wbuf[2][16][392] (async direct-to-LDS)
        for (int idx = tid; idx < 1536; idx += 256) {
            const int rr = idx / 48, c8 = idx % 48;
            const int nrow = (rr >> 4) * 192 + nt * 16 + (rr & 15);
            stage_cp16(w1b + nrow * C_DIM + c8 * 8, wbuf + rr * XN_STRIDE + c8 * 8);
        }
        stage_wait();
        __syncthreads();

        const int n0 = nh * 192 + nt * 16;
        const float bv = b_nl1[n0 + l15];
        v8f acc = { bv, bv, bv, bv, bv, bv, bv, bv };
        const __bf16* arow = Xn + (mrow + l15) * XN_STRIDE + 8 * lh;
        const __bf16* brow = wbuf + (nh * 16 + l15) * XN_STRIDE + 8 * lh;
        #pragma unroll
        for (int kt = 0; kt < 12; kt++) {
            ABf a, b;
            a.h[0] = *(const bf16x8*)(arow + kt * 32);
            a.h[1] = *(const bf16x8*)(arow + kt * 32 + 16);
            b.h[0] = *(const bf16x8*)(brow + kt * 32);
            b.h[1] = *(const bf16x8*)(brow + kt * 32 + 16);
            acc = __builtin_amdgcn_wmma_f32_16x16x32_bf16(false, a.v, false, b.v,
                                                          (short)0, acc, false, false);
        }
        #pragma unroll
        for (int j = 0; j < 8; j++) {
            float v = acc[j];
            v = v > 0.f ? v : 0.f;
            Y1[(mrow + j + 8 * lh) * XN_STRIDE + n0 + l15] = (__bf16)v;
        }
        __syncthreads();
    }

    // -------- Phase C: S = (Xn*Wl^T + bl) + (Y1*W2^T + b2), f32 in LDS --------
    for (int nt2 = 0; nt2 < 2; nt2++) {
        // stage Wl tiles for both N-halves
        for (int idx = tid; idx < 1536; idx += 256) {
            const int rr = idx / 48, c8 = idx % 48;
            const int nrow = ((rr >> 4) * 2 + nt2) * 16 + (rr & 15);
            stage_cp16(wlb + nrow * C_DIM + c8 * 8, wbuf + rr * XN_STRIDE + c8 * 8);
        }
        stage_wait();
        __syncthreads();

        const int n0 = (nh * 2 + nt2) * 16;
        const float bv = b_lin[n0 + l15] + b_nl2[n0 + l15];
        v8f acc = { bv, bv, bv, bv, bv, bv, bv, bv };
        const __bf16* arow = Xn + (mrow + l15) * XN_STRIDE + 8 * lh;
        const __bf16* brow = wbuf + (nh * 16 + l15) * XN_STRIDE + 8 * lh;
        #pragma unroll
        for (int kt = 0; kt < 12; kt++) {
            ABf a, b;
            a.h[0] = *(const bf16x8*)(arow + kt * 32);
            a.h[1] = *(const bf16x8*)(arow + kt * 32 + 16);
            b.h[0] = *(const bf16x8*)(brow + kt * 32);
            b.h[1] = *(const bf16x8*)(brow + kt * 32 + 16);
            acc = __builtin_amdgcn_wmma_f32_16x16x32_bf16(false, a.v, false, b.v,
                                                          (short)0, acc, false, false);
        }
        __syncthreads();

        // stage W2 tiles (same tile indices)
        for (int idx = tid; idx < 1536; idx += 256) {
            const int rr = idx / 48, c8 = idx % 48;
            const int nrow = ((rr >> 4) * 2 + nt2) * 16 + (rr & 15);
            stage_cp16(w2b + nrow * C_DIM + c8 * 8, wbuf + rr * XN_STRIDE + c8 * 8);
        }
        stage_wait();
        __syncthreads();

        const __bf16* arow2 = Y1 + (mrow + l15) * XN_STRIDE + 8 * lh;
        #pragma unroll
        for (int kt = 0; kt < 12; kt++) {
            ABf a, b;
            a.h[0] = *(const bf16x8*)(arow2 + kt * 32);
            a.h[1] = *(const bf16x8*)(arow2 + kt * 32 + 16);
            b.h[0] = *(const bf16x8*)(brow + kt * 32);
            b.h[1] = *(const bf16x8*)(brow + kt * 32 + 16);
            acc = __builtin_amdgcn_wmma_f32_16x16x32_bf16(false, a.v, false, b.v,
                                                          (short)0, acc, false, false);
        }
        #pragma unroll
        for (int j = 0; j < 8; j++) {
            S[(mrow + j + 8 * lh) * S_STRIDE + n0 + l15] = acc[j];
        }
        __syncthreads();
    }

    // -------- Phase D: cosine argmax + pseudo-label gather --------
    // normalize(S) is a positive per-row scaling -> cannot change argmax; skip it.
    if (tid < ROWS_PB) {
        const float* srow = S + tid * S_STRIDE;
        int best = 0;
        float bestv = -3.4e38f;
        for (int k = 0; k < K_CL; k++) {
            const float* nck = ncf + k * D_DIM;
            float d = 0.f;
            #pragma unroll
            for (int i = 0; i < D_DIM; i++) d += srow[i] * nck[i];
            if (d > bestv) { bestv = d; best = k; }
        }
        out[(size_t)blockIdx.x * ROWS_PB + tid] = pseudo[best];
    }
}

// ============================================================
extern "C" void kernel_launch(void* const* d_in, const int* in_sizes, int n_in,
                              void* d_out, int out_size, void* d_ws, size_t ws_size,
                              hipStream_t stream) {
    const float* features = (const float*)d_in[0];
    const float* w_lin    = (const float*)d_in[1];
    const float* b_lin    = (const float*)d_in[2];
    const float* w_nl1    = (const float*)d_in[3];
    const float* b_nl1    = (const float*)d_in[4];
    const float* w_nl2    = (const float*)d_in[5];
    const float* b_nl2    = (const float*)d_in[6];
    const float* centers  = (const float*)d_in[7];
    const int*   pseudo   = (const int*)d_in[8];

    char* ws = (char*)d_ws;
    __bf16* w1b = (__bf16*)(ws + WS_W1);
    __bf16* wlb = (__bf16*)(ws + WS_WL);
    __bf16* w2b = (__bf16*)(ws + WS_W2);
    float*  ncf = (float*)(ws + WS_NC);

    prep_kernel<<<769, 256, 0, stream>>>(w_nl1, w_lin, w_nl2, centers, w1b, wlb, w2b, ncf);

    const int nblocks = NPIX / ROWS_PB;  // 4096
    semantic_head_kernel<<<nblocks, 256, L_TOTAL, stream>>>(
        features, b_lin, b_nl1, b_nl2, w1b, wlb, w2b, ncf, pseudo, (int*)d_out);
}
